// GNN_12713103196622
// MI455X (gfx1250) — matrix operand order repaired and checked by reading, not compiled, not run on hardware
//
#include <hip/hip_runtime.h>
#include <hip/hip_bf16.h>
#include <math.h>

typedef __attribute__((ext_vector_type(16))) _Float16 v16h;
typedef __attribute__((ext_vector_type(8)))  _Float16 v8h;
typedef __attribute__((ext_vector_type(2)))  _Float16 v2h;
typedef __attribute__((ext_vector_type(8)))  float    v8f;

// ---------------------------------------------------------------------------
// Small utility kernels
// ---------------------------------------------------------------------------
__global__ void k_zero_f32(float* __restrict__ p, long long n) {
    long long i = blockIdx.x * (long long)blockDim.x + threadIdx.x;
    if (i < n) p[i] = 0.0f;
}

__global__ void k_deg(const int* __restrict__ dstI, float* __restrict__ deg, int E) {
    int e = blockIdx.x * blockDim.x + threadIdx.x;
    if (e < E) unsafeAtomicAdd(&deg[dstI[e]], 1.0f);
}

__global__ void k_norm(float* __restrict__ deg, int N) {
    int i = blockIdx.x * blockDim.x + threadIdx.x;
    if (i < N) deg[i] = rsqrtf(deg[i] + 1.0f);   // deg^-1/2 with self loop
}

__global__ void k_cvt_f16(const float* __restrict__ s, _Float16* __restrict__ d, int n) {
    int i = blockIdx.x * blockDim.x + threadIdx.x;
    if (i < n) d[i] = (_Float16)s[i];
}

// Collapse the two head linears: Wc = Wp1 @ Wp2 (128 x Cpad, f16), bc = bp1 @ Wp2 + bp2
__global__ void k_prep_final(const float* __restrict__ Wp1, const float* __restrict__ bp1,
                             const float* __restrict__ Wp2, const float* __restrict__ bp2,
                             _Float16* __restrict__ Wc, float* __restrict__ bc,
                             int C, int Cpad) {
    int tid = blockIdx.x * blockDim.x + threadIdx.x;
    int stride = gridDim.x * blockDim.x;
    int tot = 128 * Cpad;
    for (int idx = tid; idx < tot; idx += stride) {
        int i = idx / Cpad, j = idx % Cpad;
        float s = 0.0f;
        if (j < C) {
            for (int k = 0; k < 128; ++k) s += Wp1[i * 128 + k] * Wp2[k * C + j];
        }
        Wc[idx] = (_Float16)s;
    }
    for (int j = tid; j < Cpad; j += stride) {
        float s = 0.0f;
        if (j < C) {
            s = bp2[j];
            for (int k = 0; k < 128; ++k) s += bp1[k] * Wp2[k * C + j];
        }
        bc[j] = s;
    }
}

// ---------------------------------------------------------------------------
// WMMA GEMM:  out[r][c] = epilogue( sum_k A[r][k] * B[k][c] )
//   A: M x 128 f32 (row major), converted to f16 in registers
//   B: 128 x NCOLS f16 (row major), staged TRANSPOSED through LDS so each
//      lane's 16-half B fragment is two 16-byte runs 32 bytes apart (2x ds_load_b128)
//   epilogue: optional per-row scale, optional per-col bias, optional dup store
// One wave computes one 16-row tile across all NCOLS columns.
// ---------------------------------------------------------------------------
template <int NCOLS>
__global__ __launch_bounds__(256) void k_gemm_wmma(
    const float* __restrict__ A, const _Float16* __restrict__ B, int M,
    const float* __restrict__ rowScale, const float* __restrict__ bias,
    float* __restrict__ out, int out_ld, int store_cols,
    float* __restrict__ out2)
{
    constexpr int NT  = NCOLS / 16;
    constexpr int LDB = 136;   // padded k-stride (halfs): 272B rows, 16B-aligned frags
    __shared__ _Float16 sBt[NCOLS * LDB];

    const int tid = threadIdx.x;
    // transpose-stage: sBt[c*LDB + k] = B[k*NCOLS + c]; write K-pairs as 4B stores
    for (int p = tid; p < 64 * NCOLS; p += 256) {
        int kp = p / NCOLS;               // K pair index 0..63
        int c  = p - kp * NCOLS;          // column (coalesced across lanes)
        v2h pack;
        pack[0] = B[(2 * kp) * NCOLS + c];
        pack[1] = B[(2 * kp + 1) * NCOLS + c];
        *(v2h*)&sBt[c * LDB + 2 * kp] = pack;
    }
    __syncthreads();

    const int wave = tid >> 5;
    const int lane = tid & 31;
    const int nRowTiles = (M + 15) >> 4;
    const int tileRow = blockIdx.x * 8 + wave;
    if (tileRow >= nRowTiles) return;      // wave-uniform: EXEC stays all-1s

    const int m     = lane & 15;           // row-in-tile (A), col-in-tile (B/C/D)
    const int kbase = (lane < 16) ? 0 : 8; // A-matrix 16x32 f16 lane layout
    const int klo   = (lane < 16) ? 0 : 8; // B-matrix 32x16 f16 lane layout
    int r = tileRow * 16 + m;
    if (r >= M) r = M - 1;                 // clamp loads; stores are guarded
    const float* Arow = A + (size_t)r * 128;

    v8f acc[NT] = {};

    #pragma unroll
    for (int k0 = 0; k0 < 128; k0 += 32) {
        // A fragment: elements 0..7 <- K=k0+kbase+0..7, 8..15 <- K=k0+16+kbase+0..7
        float4 f0 = *(const float4*)(Arow + k0 + kbase);
        float4 f1 = *(const float4*)(Arow + k0 + kbase + 4);
        float4 f2 = *(const float4*)(Arow + k0 + 16 + kbase);
        float4 f3 = *(const float4*)(Arow + k0 + 16 + kbase + 4);
        v16h a;
        a[0]  = (_Float16)f0.x;  a[1]  = (_Float16)f0.y;
        a[2]  = (_Float16)f0.z;  a[3]  = (_Float16)f0.w;
        a[4]  = (_Float16)f1.x;  a[5]  = (_Float16)f1.y;
        a[6]  = (_Float16)f1.z;  a[7]  = (_Float16)f1.w;
        a[8]  = (_Float16)f2.x;  a[9]  = (_Float16)f2.y;
        a[10] = (_Float16)f2.z;  a[11] = (_Float16)f2.w;
        a[12] = (_Float16)f3.x;  a[13] = (_Float16)f3.y;
        a[14] = (_Float16)f3.z;  a[15] = (_Float16)f3.w;

        #pragma unroll
        for (int nt = 0; nt < NT; ++nt) {
            const int col = nt * 16 + m;
            // B fragment: elements 0..7 <- K=k0+klo+0..7, 8..15 <- K=k0+16+klo+0..7 (16B-aligned)
            const _Float16* bp = &sBt[col * LDB + k0 + klo];
            v8h blo = *(const v8h*)bp;
            v8h bhi = *(const v8h*)(bp + 16);
            v16h b = __builtin_shufflevector(blo, bhi,
                0, 1, 2, 3, 4, 5, 6, 7, 8, 9, 10, 11, 12, 13, 14, 15);
            acc[nt] = __builtin_amdgcn_wmma_f32_16x16x32_f16(
                false, a, false, b, (short)0, acc[nt], false, false);
        }
    }

    // C/D layout: VGPR j -> row mOff+j, col = lane%16
    const int mOff = (lane < 16) ? 0 : 8;
    #pragma unroll
    for (int nt = 0; nt < NT; ++nt) {
        #pragma unroll
        for (int j = 0; j < 8; ++j) {
            int rr = tileRow * 16 + mOff + j;
            int cc = nt * 16 + m;
            if (rr < M && cc < store_cols) {
                float val = acc[nt][j];
                if (rowScale) val *= rowScale[rr];
                if (bias)     val += bias[cc];
                out[(size_t)rr * out_ld + cc] = val;
                if (out2) out2[(size_t)rr * out_ld + cc] = val;  // agg = g (self-loop)
            }
        }
    }
}

// ---------------------------------------------------------------------------
// Edge scatter: agg[dst] += g[src]   (128 floats per edge, 4 per thread)
// g rows are L2-resident (51MB << 192MB L2) -> hardware f32 atomics at L2.
// ---------------------------------------------------------------------------
__global__ void k_scatter(const int* __restrict__ srcI, const int* __restrict__ dstI,
                          const float* __restrict__ g, float* __restrict__ agg, int E) {
    long long tid = blockIdx.x * (long long)blockDim.x + threadIdx.x;
    long long total = (long long)E * 32;
    if (tid >= total) return;
    int e = (int)(tid >> 5);
    int c = ((int)tid & 31) * 4;
    int s = srcI[e], d = dstI[e];
    const float4 v = *(const float4*)(g + (size_t)s * 128 + c);
    float* p = agg + (size_t)d * 128 + c;
    unsafeAtomicAdd(p + 0, v.x);
    unsafeAtomicAdd(p + 1, v.y);
    unsafeAtomicAdd(p + 2, v.z);
    unsafeAtomicAdd(p + 3, v.w);
}

// h = relu(norm[r] * agg + bias[c])
__global__ void k_epilogue(const float* __restrict__ agg, const float* __restrict__ norm,
                           const float* __restrict__ bias, float* __restrict__ h, int N) {
    long long i = blockIdx.x * (long long)blockDim.x + threadIdx.x;
    long long tot = (long long)N * 128;
    if (i >= tot) return;
    int r = (int)(i >> 7);
    int c = (int)i & 127;
    float v = norm[r] * agg[i] + bias[c];
    h[i] = v > 0.0f ? v : 0.0f;
}

// Row-wise log_softmax over C columns; one wave32 per row, shuffle reductions.
__global__ __launch_bounds__(256) void k_logsoftmax(const float* __restrict__ logits,
                                                    float* __restrict__ out, int N, int C) {
    int wave = threadIdx.x >> 5, lane = threadIdx.x & 31;
    int row = blockIdx.x * 8 + wave;
    if (row >= N) return;
    const float* p = logits + (size_t)row * C;
    float v0 = (lane < C) ? p[lane] : -__builtin_inff();
    float v1 = (lane + 32 < C) ? p[lane + 32] : -__builtin_inff();
    float mx = fmaxf(v0, v1);
    #pragma unroll
    for (int o = 16; o > 0; o >>= 1) mx = fmaxf(mx, __shfl_xor(mx, o, 32));
    float s = 0.0f;
    if (lane < C)      s += expf(v0 - mx);
    if (lane + 32 < C) s += expf(v1 - mx);
    #pragma unroll
    for (int o = 16; o > 0; o >>= 1) s += __shfl_xor(s, o, 32);
    float ls = logf(s);
    float* q = out + (size_t)row * C;
    if (lane < C)      q[lane]      = v0 - mx - ls;
    if (lane + 32 < C) q[lane + 32] = v1 - mx - ls;
}

// ---------------------------------------------------------------------------
extern "C" void kernel_launch(void* const* d_in, const int* in_sizes, int n_in,
                              void* d_out, int out_size, void* d_ws, size_t ws_size,
                              hipStream_t stream) {
    const float* x   = (const float*)d_in[0];
    const int*   ei  = (const int*)d_in[1];
    const float* W1  = (const float*)d_in[2];
    const float* b1  = (const float*)d_in[3];
    const float* W2  = (const float*)d_in[4];
    const float* b2  = (const float*)d_in[5];
    const float* W3  = (const float*)d_in[6];
    const float* b3  = (const float*)d_in[7];
    const float* Wp1 = (const float*)d_in[8];
    const float* bp1 = (const float*)d_in[9];
    const float* Wp2 = (const float*)d_in[10];
    const float* bp2 = (const float*)d_in[11];

    const int Nn = in_sizes[0] / 128;   // nodes
    const int Ee = in_sizes[1] / 2;     // edges
    const int Cc = in_sizes[11];        // classes (40)
    const int Cpad = 48;                // 3 WMMA N-tiles
    const int* srcI = ei;
    const int* dstI = ei + Ee;

    // workspace carve-up (256B aligned)
    size_t off = 0;
    auto take = [&](size_t bytes) {
        size_t o = off;
        off += (bytes + 255) & ~(size_t)255;
        return o;
    };
    char* ws = (char*)d_ws;
    float*    norm  = (float*)   (ws + take((size_t)Nn * 4));
    _Float16* Wf16  = (_Float16*)(ws + take((size_t)3 * 128 * 128 * 2));
    _Float16* Wc    = (_Float16*)(ws + take((size_t)128 * Cpad * 2));
    float*    bc    = (float*)   (ws + take((size_t)Cpad * 4));
    float*    bufA  = (float*)   (ws + take((size_t)Nn * 128 * 4));  // g / logits
    float*    bufB  = (float*)   (ws + take((size_t)Nn * 128 * 4));  // agg
    float*    bufC  = (float*)   (ws + take((size_t)Nn * 128 * 4));  // h

    const int T = 256;
    // degrees -> norm = rsqrt(deg+1)
    k_zero_f32<<<(Nn + T - 1) / T, T, 0, stream>>>(norm, Nn);
    k_deg<<<(Ee + T - 1) / T, T, 0, stream>>>(dstI, norm, Ee);
    k_norm<<<(Nn + T - 1) / T, T, 0, stream>>>(norm, Nn);

    // weights to f16; fold head linears
    k_cvt_f16<<<64, T, 0, stream>>>(W1, Wf16 + 0 * 16384, 16384);
    k_cvt_f16<<<64, T, 0, stream>>>(W2, Wf16 + 1 * 16384, 16384);
    k_cvt_f16<<<64, T, 0, stream>>>(W3, Wf16 + 2 * 16384, 16384);
    k_prep_final<<<32, T, 0, stream>>>(Wp1, bp1, Wp2, bp2, Wc, bc, Cc, Cpad);

    const int nRowTiles = (Nn + 15) / 16;
    const int gemmBlocks = (nRowTiles + 7) / 8;
    const long long scatThreads = (long long)Ee * 32;
    const int scatBlocks = (int)((scatThreads + T - 1) / T);
    const int epiBlocks = (int)(((long long)Nn * 128 + T - 1) / T);

    const float* bs[3] = {b1, b2, b3};
    const float* hin = x;
    for (int l = 0; l < 3; ++l) {
        // g = norm ⊙ (h @ W); also agg := g (self-loop term)
        k_gemm_wmma<128><<<gemmBlocks, T, 0, stream>>>(
            hin, Wf16 + (size_t)l * 16384, Nn, norm, nullptr,
            bufA, 128, 128, bufB);
        // agg[dst] += g[src]
        k_scatter<<<scatBlocks, T, 0, stream>>>(srcI, dstI, bufA, bufB, Ee);
        // h = relu(norm ⊙ agg + b)
        k_epilogue<<<epiBlocks, T, 0, stream>>>(bufB, norm, bs[l], bufC, Nn);
        hin = bufC;
    }

    // logits = h3 @ Wc + bc   (head collapsed to one GEMM), then log_softmax
    k_gemm_wmma<48><<<gemmBlocks, T, 0, stream>>>(
        bufC, Wc, Nn, nullptr, bc, bufA, Cc, Cc, nullptr);
    k_logsoftmax<<<(Nn + 7) / 8, T, 0, stream>>>(bufA, (float*)d_out, Nn, Cc);
}
